// ConcentrationPredictor_60129542276
// MI455X (gfx1250) — matrix operand order repaired and checked by reading, not compiled, not run on hardware
//
#include <hip/hip_runtime.h>
#include <math.h>

// CDNA5 / gfx1250: wave32, WMMA 16x16x32 f16->f32.
typedef __attribute__((ext_vector_type(16))) _Float16 v16h;
typedef __attribute__((ext_vector_type(8)))  float    v8f;

#define TILE     512
#define HALO     128              // 4 stages/step * 32 steps
#define MAXL     (TILE + 2*HALO)  // 768 points per workgroup working set
#define NTHREADS 256              // 8 waves (wave32)

// Hardware exp2 (v_exp_f32), branch-free.
__device__ __forceinline__ float exp2_hw(float x) {
#if __has_builtin(__builtin_amdgcn_exp2f)
    return __builtin_amdgcn_exp2f(x);
#else
    return exp2f(x);
#endif
}

// Branch-free tanh: 1 - 2/(exp2(2*log2e*x) + 1).
// Clamp the exponent so exp2 never produces inf (rcp then gives exact +/-1).
__device__ __forceinline__ float fast_tanh(float x) {
    float e = exp2_hw(fminf(x * 2.885390082f, 126.0f));   // exp(2x)
    return 1.0f - 2.0f * __builtin_amdgcn_rcpf(e + 1.0f);
}

// Branch-free sigmoid: 1/(1 + exp(-x)).
__device__ __forceinline__ float fast_sigmoid(float x) {
    float e = exp2_hw(fminf(-x * 1.442695041f, 126.0f));  // exp(-x)
    return __builtin_amdgcn_rcpf(1.0f + e);
}

__global__ __launch_bounds__(NTHREADS) void rk4_mlp_diffusion(
    const float* __restrict__ c0, const float* __restrict__ t,
    const float* __restrict__ W1, const float* __restrict__ b1,
    const float* __restrict__ W2, const float* __restrict__ b2,
    const float* __restrict__ W3, const float* __restrict__ b3,
    const float* __restrict__ W4, const float* __restrict__ b4,
    const float* __restrict__ p_exp,
    float* __restrict__ out, int N, int T)
{
    __shared__ float sC[MAXL];    // current state c
    __shared__ float sYa[MAXL];   // RK stage buffer A
    __shared__ float sYb[MAXL];   // RK stage buffer B
    __shared__ float sAcc[MAXL];  // k1 + 2k2 + 2k3 + k4 accumulator

    const float DXc  = 0.04f;
    const float D0   = 0.0005f / (DXc * DXc);   // 0.3125
    const float BC00 = 1.0f;

    const int tid    = threadIdx.x;
    const int wv     = tid >> 5;          // wave id within block
    const int lane   = tid & 31;
    const bool hiH   = lane >= 16;        // high half of wave
    const int m      = lane & 15;         // matrix row / point-in-batch
    const int kb     = hiH ? 8 : 0;       // C/D row (output-feature) base per lane half

    const int base = blockIdx.x * TILE;
    const int lo   = max(0, base - HALO);
    const int hicl = min(N, base + TILE + HALO);
    const int L    = hicl - lo;

    // Load tile (+halo) of initial state into LDS; emit output row 0.
    for (int j = tid; j < L; j += NTHREADS) sC[j] = c0[lo + j];
    for (int j = tid; j < TILE && base + j < N; j += NTHREADS)
        out[base + j] = c0[base + j];

    const float scale = exp2_hw(p_exp[0] * 3.321928095f);  // 10^p_exp

    // --- Constant WMMA A fragments: A = W^T (outfeat x infeat), 16x32 f16 ---
    // A operand in this code's contiguous k order: lane holds row M=lane&15; lanes 0-15 hold
    // K = 0..15 in elements 0..15; lanes 16-31 would be K=16..31 -> zero
    // padding (K is only 15 wide).
    v16h A2 = {};
    v16h A3 = {};
#pragma unroll
    for (int j = 0; j < 16; ++j) {
        int k = hiH ? 15 : j;                             // input feature
        float w2v = (k < 15 && m < 15) ? W2[k * 15 + m] : 0.0f;  // W2[in][out]
        float w3v = (k < 15 && m < 15) ? W3[k * 15 + m] : 0.0f;
        A2[j] = (_Float16)w2v;
        A3[j] = (_Float16)w3v;
    }
    // Bias as C accumulator seed: C/D layout lane holds col N=lane&15 (point),
    // rows M = v + 8*(lane>=16) (output feature) across 8 VGPRs.
    v8f Cb2, Cb3;
    float w4s[8];
#pragma unroll
    for (int v = 0; v < 8; ++v) {
        int f = kb + v;
        Cb2[v]  = (f < 15) ? b2[f] : 0.0f;
        Cb3[v]  = (f < 15) ? b3[f] : 0.0f;
        w4s[v]  = (f < 15) ? W4[f] : 0.0f;
    }
    const float b4s = b4[0];
    float w1r[15], b1r[15];
#pragma unroll
    for (int k = 0; k < 15; ++k) { w1r[k] = W1[k]; b1r[k] = b1[k]; }

    __syncthreads();

    const int steps = T - 1;
    const int nb    = (L + 15) >> 4;   // 16-point batches per stage

    for (int s = 0; s < steps; ++s) {
        const float dt = t[s + 1] - t[s];

        for (int stage = 0; stage < 4; ++stage) {
            // stage inputs:  c, yA, yB, yA ; stage outputs: yA, yB, yA, (none)
            const float* yin  = (stage == 0) ? sC : ((stage == 2) ? sYb : sYa);
            float*       yout = (stage == 1) ? sYb : sYa;
            const float  aw   = (stage == 1 || stage == 2) ? 2.0f : 1.0f;
            const float  yc   = (stage == 2) ? dt : 0.5f * dt;

            for (int b = wv; b < nb; b += NTHREADS / 32) {
                const int p  = (b << 4) + m;          // local point index
                const int pc = (p < L) ? p : (L - 1);
                const float cv = yin[pc];

                // ---- Layer 1 (fan-in 1): build B fragment of H1^T ----
                // B operand in contiguous k order (as A): lanes 0-15 hold column N=lane
                // with K=0..15 in elements 0..15; lanes 16-31 hold K=16..31
                // (all zero pad here).
                v16h B1 = {};
                if (!hiH) {
                    const float x = cv * scale;
#pragma unroll
                    for (int k = 0; k < 15; ++k)
                        B1[k] = (_Float16)fast_tanh(fmaf(x, w1r[k], b1r[k]));
                }

                // ---- Layer 2: H2^T = tanh(W2^T * H1^T + b2) via WMMA ----
                v8f acc2 = Cb2;
                acc2 = __builtin_amdgcn_wmma_f32_16x16x32_f16(
                           false, A2, false, B1, (short)0, acc2, false, false);

                // tanh, f32->f16 pack, and cross-half exchange to rebuild a
                // B fragment (lane<16 needs feats 8..15 from lane+16).
                unsigned pk[4];
#pragma unroll
                for (int v = 0; v < 4; ++v) {
                    union { _Float16 h[2]; unsigned u; } cvt;
                    cvt.h[0] = (_Float16)fast_tanh(acc2[2 * v]);
                    cvt.h[1] = (_Float16)fast_tanh(acc2[2 * v + 1]);
                    pk[v] = cvt.u;
                }
                unsigned ex[4];
#pragma unroll
                for (int v = 0; v < 4; ++v)
                    ex[v] = (unsigned)__shfl_xor((int)pk[v], 16, 32);

                v16h B2 = {};
                if (!hiH) {
#pragma unroll
                    for (int v = 0; v < 4; ++v) {
                        union { unsigned u; _Float16 h[2]; } a, bq;
                        a.u = pk[v]; bq.u = ex[v];
                        B2[2 * v]     = a.h[0];
                        B2[2 * v + 1] = a.h[1];
                        B2[8 + 2 * v]     = bq.h[0];
                        B2[8 + 2 * v + 1] = bq.h[1];
                    }
                }

                // ---- Layer 3: H3^T = tanh(W3^T * H2^T + b3) via WMMA ----
                v8f acc3 = Cb3;
                acc3 = __builtin_amdgcn_wmma_f32_16x16x32_f16(
                           false, A3, false, B2, (short)0, acc3, false, false);

                // ---- Layer 4 (fan-out 1): split dot + cross-half reduce ----
                float part = 0.0f;
#pragma unroll
                for (int v = 0; v < 8; ++v)
                    part = fmaf(fast_tanh(acc3[v]), w4s[v], part);
                part += __shfl_xor(part, 16, 32);
                const float retv = fast_sigmoid(part + b4s);

                // ---- Stencil flux + RK4 bookkeeping (one lane per point) ----
                if (!hiH && p < L) {
                    const int   g  = lo + p;
                    const float cc = cv;
                    const float cm = yin[(p > 0) ? p - 1 : 0];
                    const float cp = yin[(p < L - 1) ? p + 1 : L - 1];

                    float left = (g == 0)
                        ? D0 * retv * (BC00 - cc)
                        : D0 * retv * (cm - cc);
                    float right;
                    if (g == N - 1) {
                        const float rbc = D0 * DXc * (cm - cc);
                        right = D0 * retv * (rbc - cc);
                    } else {
                        right = D0 * retv * (cp - cc);
                    }
                    const float kk = left + right;

                    if (stage == 0) sAcc[p] = kk;
                    else            sAcc[p] += aw * kk;
                    if (stage < 3)  yout[p] = sC[p] + yc * kk;
                }
            }
            __syncthreads();
        }

        // c += dt/6 * (k1 + 2k2 + 2k3 + k4)
        for (int j = tid; j < L; j += NTHREADS)
            sC[j] += (dt * (1.0f / 6.0f)) * sAcc[j];
        __syncthreads();

        // Emit output row s+1 (interior slice only; halo is redundant work).
        const size_t row = (size_t)(s + 1) * (size_t)N;
        const int    off = base - lo;  // 0 for block 0, HALO otherwise
        for (int j = tid; j < TILE && base + j < N; j += NTHREADS)
            out[row + base + j] = sC[off + j];
        // No barrier needed: next step's stage 0 only reads sC, writes sYa/sAcc.
    }
}

extern "C" void kernel_launch(void* const* d_in, const int* in_sizes, int n_in,
                              void* d_out, int out_size, void* d_ws, size_t ws_size,
                              hipStream_t stream) {
    const float* c0    = (const float*)d_in[0];
    const float* t     = (const float*)d_in[1];
    const float* W1    = (const float*)d_in[2];
    const float* b1    = (const float*)d_in[3];
    const float* W2    = (const float*)d_in[4];
    const float* b2    = (const float*)d_in[5];
    const float* W3    = (const float*)d_in[6];
    const float* b3    = (const float*)d_in[7];
    const float* W4    = (const float*)d_in[8];
    const float* b4    = (const float*)d_in[9];
    const float* p_exp = (const float*)d_in[10];

    const int N = in_sizes[0];   // 65536
    const int T = in_sizes[1];   // 33

    const int grid = (N + TILE - 1) / TILE;   // 128 workgroups
    rk4_mlp_diffusion<<<grid, NTHREADS, 0, stream>>>(
        c0, t, W1, b1, W2, b2, W3, b3, W4, b4, p_exp,
        (float*)d_out, N, T);
}
